// vsgnet_82600811036872
// MI455X (gfx1250) — matrix operand order repaired and checked by reading, not compiled, not run on hardware
//
#include <hip/hip_runtime.h>
#include <hip/hip_bf16.h>

typedef __attribute__((ext_vector_type(16))) __bf16 v16bf;
typedef __attribute__((ext_vector_type(8)))  float  v8f;
typedef __attribute__((ext_vector_type(4)))  int    v4i;

struct alignas(16) U4 { unsigned int x, y, z, w; };
union Frag { v16bf v; U4 q[2]; };

struct F4 { float x, y, z, w; };

#define DD 512
#define HH 512
#define CC 16
#define NROWS 32768   // B*R = 64*512

#define AS1 __attribute__((address_space(1)))
#define AS3 __attribute__((address_space(3)))

#if __has_builtin(__builtin_amdgcn_global_load_async_to_lds_b128)
#define VSG_ASYNC_LDS 1
#else
#define VSG_ASYNC_LDS 0
#endif

__device__ __forceinline__ unsigned short f32_to_bf16(float f) {
    union { float f; unsigned u; } v; v.f = f;
    unsigned lsb = (v.u >> 16) & 1u;
    unsigned r = v.u + 0x7FFFu + lsb;   // round-to-nearest-even
    return (unsigned short)(r >> 16);
}

__device__ __forceinline__ void vsg_wait_async0() {
#if VSG_ASYNC_LDS
#if __has_builtin(__builtin_amdgcn_s_wait_asynccnt)
    __builtin_amdgcn_s_wait_asynccnt(0);
#else
    asm volatile("s_wait_asynccnt 0x0" ::: "memory");
#endif
#endif
}

#if VSG_ASYNC_LDS
// Each thread moves 64B of the 16KB W1T block: 4 async b128 lane-copies.
__device__ __forceinline__ void vsg_issue_w1_async(const unsigned short* gsrc,
                                                   unsigned short* ldst, int tid) {
    #pragma unroll
    for (int t = 0; t < 4; ++t) {
        int off = (tid + t * 256) * 8;   // in halves: 8 halves = 16B per lane
        __builtin_amdgcn_global_load_async_to_lds_b128(
            (AS1 v4i*)(gsrc + off), (AS3 v4i*)(ldst + off), 0, 0);
    }
}
#endif

// ---------------------------------------------------------------------------
// Prep 1: build bf16 activation matrices  Xb[3][NROWS][512]
//   br0 = spatial, br1 = f_oo * spatial, br2 = 0.5*(G[i0]+G[i1]) gathered
// ---------------------------------------------------------------------------
__global__ void prep_x(const float* __restrict__ g, const float* __restrict__ s,
                       const float* __restrict__ fv, const int* __restrict__ pairs,
                       unsigned short* __restrict__ Xb)
{
    int row = blockIdx.x;          // 0 .. 3*NROWS-1
    int br  = row >> 15;
    int n   = row & 32767;
    int d   = threadIdx.x * 4;     // 128 threads * 4 floats = 512
    size_t off = (size_t)n * DD + d;
    float v0, v1, v2, v3;
    if (br == 0) {
        F4 a = *(const F4*)(s + off);
        v0 = a.x; v1 = a.y; v2 = a.z; v3 = a.w;
    } else if (br == 1) {
        F4 a = *(const F4*)(s + off);
        F4 c = *(const F4*)(fv + off);
        v0 = a.x * c.x; v1 = a.y * c.y; v2 = a.z * c.z; v3 = a.w * c.w;
    } else {
        int b  = n >> 9;                       // R = 512 pairs per batch
        int i0 = pairs[2 * n];
        int i1 = pairs[2 * n + 1];
        F4 a = *(const F4*)(g + ((size_t)(b * 64 + i0) * DD + d));
        F4 c = *(const F4*)(g + ((size_t)(b * 64 + i1) * DD + d));
        v0 = 0.5f * (a.x + c.x); v1 = 0.5f * (a.y + c.y);
        v2 = 0.5f * (a.z + c.z); v3 = 0.5f * (a.w + c.w);
    }
    unsigned short* o = Xb + (size_t)row * DD + d;
    o[0] = f32_to_bf16(v0); o[1] = f32_to_bf16(v1);
    o[2] = f32_to_bf16(v2); o[3] = f32_to_bf16(v3);
}

// ---------------------------------------------------------------------------
// Prep 2: W1 [K,D,H] fp32 (per branch) -> W1T[b][k][h][d] bf16 (transposed)
// ---------------------------------------------------------------------------
__global__ void prep_w1(const float* __restrict__ w0, const float* __restrict__ w1,
                        const float* __restrict__ w2, unsigned short* __restrict__ W1T)
{
    size_t idx = (size_t)blockIdx.x * 256 + threadIdx.x;   // < 9*512*512
    int d  = (int)(idx & 511);
    int h  = (int)((idx >> 9) & 511);
    int kb = (int)(idx >> 18);          // b*3 + k
    int b  = kb / 3, k = kb % 3;
    const float* w = (b == 0) ? w0 : (b == 1) ? w1 : w2;
    float v = w[((size_t)k * DD + d) * HH + h];
    W1T[idx] = f32_to_bf16(v);
}

// ---------------------------------------------------------------------------
// Prep 3: W2 [K,H,C] fp32 (per branch) -> W2T[b][k][c][h] bf16 (transposed)
// ---------------------------------------------------------------------------
__global__ void prep_w2(const float* __restrict__ w0, const float* __restrict__ w1,
                        const float* __restrict__ w2, unsigned short* __restrict__ W2T)
{
    size_t idx = (size_t)blockIdx.x * 256 + threadIdx.x;   // < 9*16*512
    int h  = (int)(idx & 511);
    int c  = (int)((idx >> 9) & 15);
    int kb = (int)(idx >> 13);          // b*3 + k
    int b  = kb / 3, k = kb % 3;
    const float* w = (b == 0) ? w0 : (b == 1) ? w1 : w2;
    float v = w[((size_t)k * HH + h) * CC + c];
    W2T[idx] = f32_to_bf16(v);
}

// ---------------------------------------------------------------------------
// Main fused kernel: for each (k, 128-row tile):
//   for branch in {spatial, refined, graphical}:
//     h = relu(X @ W1[k] + b1[k]);  y = sigmoid(h @ W2[k] + b2[k]); prod *= y
// 8 waves per WG; each wave owns 16 rows (A-fragments pinned in VGPRs).
// W1T slices stream through a double-buffered LDS tile via ASYNC global->LDS
// DMA (no VGPR staging, no spills): per iteration
//   s_wait_asynccnt 0 ; barrier ; issue async block j+1 ; 16 WMMAs on block j
// ---------------------------------------------------------------------------
__global__ __launch_bounds__(256)
void vsg_main(const unsigned short* __restrict__ Xb,
              const unsigned short* __restrict__ W1T,
              const unsigned short* __restrict__ W2T,
              const float* __restrict__ b1_0, const float* __restrict__ b2_0,
              const float* __restrict__ b1_1, const float* __restrict__ b2_1,
              const float* __restrict__ b1_2, const float* __restrict__ b2_2,
              float* __restrict__ out)
{
    __shared__ __align__(16) unsigned short w1blk[2][16 * DD];  // 2 x 16 KB
    __shared__ __align__(16) unsigned short hstage[8][16 * 32]; // 8 KB per-wave h tiles

    const int k    = blockIdx.y;              // relation key 0..2
    const int lane = threadIdx.x & 31;
    const int wv   = threadIdx.x >> 5;        // 0..7
    const int col  = lane & 15;               // N for B/C/D, M-row for A
    const int hi   = lane >> 4;               // lane-half select
    const int rb   = blockIdx.x * 128 + wv * 16;

    const float* b1p[3] = { b1_0, b1_1, b1_2 };
    const float* b2p[3] = { b2_0, b2_1, b2_2 };

    v8f prod;
    #pragma unroll
    for (int f = 0; f < 8; ++f) prod[f] = 1.0f;

    #pragma unroll
    for (int br = 0; br < 3; ++br) {
        // ---- load this wave's 16 A-fragments (16 rows x 512 of D) ----
        Frag A[16];
        const unsigned short* xrow = Xb + ((size_t)br * NROWS + rb + col) * DD;
        #pragma unroll
        for (int dc = 0; dc < 16; ++dc) {
            int kb = dc * 32 + hi * 8;
            A[dc].q[0] = *(const U4*)(xrow + kb);
            A[dc].q[1] = *(const U4*)(xrow + kb + 16);
        }

        const unsigned short* w1base = W1T + ((size_t)br * 3 + k) * (DD * HH);
        const unsigned short* w2base = W2T + ((size_t)br * 3 + k) * (CC * HH);
        const float* b1 = b1p[br] + k * HH;
        const float  b2v = b2p[br][k * CC + col];

        v8f acc2;
        #pragma unroll
        for (int f = 0; f < 8; ++f) acc2[f] = 0.0f;

#if VSG_ASYNC_LDS
        // pipeline prologue: start async DMA of W1T block 0 into buffer 0
        vsg_issue_w1_async(w1base, &w1blk[0][0], threadIdx.x);
#endif

        for (int j = 0; j < 32; ++j) {
#if VSG_ASYNC_LDS
            vsg_wait_async0();     // own async stores for block j complete
            __syncthreads();       // all waves: block j resident; j-1 reads done
            if (j < 31)            // start DMA of block j+1 into the other buffer
                vsg_issue_w1_async(w1base + (size_t)(j + 1) * 16 * DD,
                                   &w1blk[(j + 1) & 1][0], threadIdx.x);
            const unsigned short* blk = w1blk[j & 1];
#else
            // fallback: plain cooperative copy (no register pipeline -> no spill)
            __syncthreads();
            {
                const U4* src = (const U4*)(w1base + (size_t)j * 16 * DD);
                U4* dst = (U4*)w1blk[0];
                #pragma unroll
                for (int t = 0; t < 4; ++t)
                    dst[threadIdx.x + t * 256] = src[threadIdx.x + t * 256];
            }
            __syncthreads();
            const unsigned short* blk = w1blk[0];
#endif

            // GEMM1 tile: acch[16x16] = X[16x512] * W1[512, j*16..+15]
            v8f acch;
            #pragma unroll
            for (int f = 0; f < 8; ++f) acch[f] = 0.0f;
            #pragma unroll
            for (int dc = 0; dc < 16; ++dc) {
                Frag Bf;
                const unsigned short* wp = blk + col * DD + dc * 32 + hi * 8;
                Bf.q[0] = *(const U4*)(wp);
                Bf.q[1] = *(const U4*)(wp + 16);
                acch = __builtin_amdgcn_wmma_f32_16x16x32_bf16(
                           false, A[dc].v, false, Bf.v, (short)0, acch, false, false);
            }

            // bias + ReLU, stage as bf16 into this wave's LDS h-tile
            float b1v = b1[j * 16 + col];
            unsigned short* hst = &hstage[wv][0];
            #pragma unroll
            for (int f = 0; f < 8; ++f) {
                float v = acch[f] + b1v;
                v = v > 0.0f ? v : 0.0f;
                int row = f + hi * 8;
                hst[row * 32 + (j & 1) * 16 + col] = f32_to_bf16(v);
            }

            // every second slice: fold 32 H-columns into GEMM2
            if (j & 1) {
                Frag A2, B2;
                const unsigned short* hp = hst + col * 32 + hi * 8;
                A2.q[0] = *(const U4*)(hp);
                A2.q[1] = *(const U4*)(hp + 16);
                const unsigned short* w2p = w2base + col * HH + (j - 1) * 16 + hi * 8;
                B2.q[0] = *(const U4*)(w2p);
                B2.q[1] = *(const U4*)(w2p + 16);
                acc2 = __builtin_amdgcn_wmma_f32_16x16x32_bf16(
                           false, A2.v, false, B2.v, (short)0, acc2, false, false);
            }
        }

        // bias + sigmoid; accumulate 3-branch product
        #pragma unroll
        for (int f = 0; f < 8; ++f) {
            float v = acc2[f] + b2v;
            prod[f] *= 1.0f / (1.0f + __expf(-v));
        }
    }

    // write out [K][NROWS][C]
    #pragma unroll
    for (int f = 0; f < 8; ++f) {
        int n = rb + f + hi * 8;
        out[((size_t)k * NROWS + n) * CC + col] = prod[f];
    }
}

// ---------------------------------------------------------------------------
extern "C" void kernel_launch(void* const* d_in, const int* in_sizes, int n_in,
                              void* d_out, int out_size, void* d_ws, size_t ws_size,
                              hipStream_t stream)
{
    const float* g  = (const float*)d_in[0];   // graphical_branch_output [4096,512]
    const float* s  = (const float*)d_in[1];   // spatial_branch_output  [32768,512]
    const float* fv = (const float*)d_in[2];   // f_oo_vis               [32768,512]
    const int*   pr = (const int*)d_in[3];     // obj_pairs              [64,512,2]

    const float* W1b[3] = { (const float*)d_in[4],  (const float*)d_in[8],  (const float*)d_in[12] };
    const float* b1b[3] = { (const float*)d_in[5],  (const float*)d_in[9],  (const float*)d_in[13] };
    const float* W2b[3] = { (const float*)d_in[6],  (const float*)d_in[10], (const float*)d_in[14] };
    const float* b2b[3] = { (const float*)d_in[7],  (const float*)d_in[11], (const float*)d_in[15] };

    // workspace partition (bf16 halves)
    unsigned short* Xb  = (unsigned short*)d_ws;             // 3*32768*512
    unsigned short* W1T = Xb + (size_t)3 * NROWS * DD;       // 9*512*512
    unsigned short* W2T = W1T + (size_t)9 * DD * HH;         // 9*16*512

    prep_x <<<dim3(3 * NROWS), 128, 0, stream>>>(g, s, fv, pr, Xb);
    prep_w1<<<dim3(9 * DD * HH / 256), 256, 0, stream>>>(W1b[0], W1b[1], W1b[2], W1T);
    prep_w2<<<dim3(9 * CC * HH / 256), 256, 0, stream>>>(W2b[0], W2b[1], W2b[2], W2T);

    vsg_main<<<dim3(NROWS / 128, 3), 256, 0, stream>>>(
        Xb, W1T, W2T,
        b1b[0], b2b[0], b1b[1], b2b[1], b1b[2], b2b[2],
        (float*)d_out);
}
